// EmbedAlignModel_68049461837851
// MI455X (gfx1250) — matrix operand order repaired and checked by reading, not compiled, not run on hardware
//
#include <hip/hip_runtime.h>
#include <hip/hip_bf16.h>
#include <math.h>

typedef __attribute__((ext_vector_type(16))) _Float16 v16h;
typedef __attribute__((ext_vector_type(8)))  float    v8f;

#define DD    256
#define LXN   256
#define NNEG  8192
#define G4D   1024

// ---------------------------------------------------------------------------
// WMMA fragment loaders (ISA 7.12.2 layouts, wave32)
// A 16x32 f16: lane l -> M = l&15, half = l>>4.
//   VGPR i in 0..3: K = 2i,2i+1 (+8*half); VGPR 4..7: K = 16+2(i-4),+1 (+8*half)
// B 32x16 f16: lane l -> N = l&15, half = l>>4. VGPR i: same K as A (2i,+1 or 16+2(i-4),+1; +8*half)
// C  f32 16x16: VGPR r -> M = r + 8*half, N = l&15
// ---------------------------------------------------------------------------
__device__ __forceinline__ v16h frag_a_f32(const float* __restrict__ arow, int ks, int half) {
    v16h a;
#pragma unroll
    for (int i = 0; i < 8; ++i) {
        int k = ks + ((i < 4) ? 2 * i : 16 + 2 * (i - 4)) + 8 * half;
        a[2 * i]     = (_Float16)arow[k];
        a[2 * i + 1] = (_Float16)arow[k + 1];
    }
    return a;
}
__device__ __forceinline__ v16h frag_a_f16(const _Float16* __restrict__ arow, int ks, int half) {
    v16h a;
#pragma unroll
    for (int i = 0; i < 8; ++i) {
        int k = ks + ((i < 4) ? 2 * i : 16 + 2 * (i - 4)) + 8 * half;
        a[2 * i]     = arow[k];
        a[2 * i + 1] = arow[k + 1];
    }
    return a;
}
__device__ __forceinline__ v16h frag_b_f32(const float* __restrict__ bcol, int ks, int half) {
    v16h b;
#pragma unroll
    for (int i = 0; i < 8; ++i) {
        int k = ks + ((i < 4) ? 2 * i : 16 + 2 * (i - 4)) + 8 * half;
        b[2 * i]     = (_Float16)bcol[k];
        b[2 * i + 1] = (_Float16)bcol[k + 1];
    }
    return b;
}

__device__ __forceinline__ float sigmoidf_(float x) { return 1.0f / (1.0f + expf(-x)); }

// ---------------------------------------------------------------------------
// 0) zero accumulators / denominators in workspace
// ---------------------------------------------------------------------------
__global__ void ea_init(float* __restrict__ den_x, float* __restrict__ den_y,
                        float* __restrict__ acc) {
    int i = threadIdx.x;
    den_x[i] = 0.0f;
    den_y[i] = 0.0f;
    if (i < 16) acc[i] = 0.0f;
}

// ---------------------------------------------------------------------------
// 1) Gx = E[idx] @ W^T + b1 + b2   (M=256, N=1024, K=256), one wave per tile
// ---------------------------------------------------------------------------
__global__ void ea_gx_gemm(const float* __restrict__ E, const int* __restrict__ idx,
                           const float* __restrict__ W, const float* __restrict__ b1,
                           const float* __restrict__ b2, float* __restrict__ out) {
    const int l = threadIdx.x;
    const int m = l & 15, half = l >> 4;
    const float* arow = E + (size_t)idx[blockIdx.x * 16 + m] * DD;
    const float* bcol = W + (size_t)(blockIdx.y * 16 + m) * DD;   // B[k][n] = W[n][k]
    v8f c = {};
#pragma unroll
    for (int ks = 0; ks < DD; ks += 32) {
        v16h a = frag_a_f32(arow, ks, half);
        v16h b = frag_b_f32(bcol, ks, half);
        c = __builtin_amdgcn_wmma_f32_16x16x32_f16(false, a, false, b, (short)0, c, false, false);
    }
#pragma unroll
    for (int r = 0; r < 8; ++r) {
        int orow = blockIdx.x * 16 + r + 8 * half;
        int ocol = blockIdx.y * 16 + m;
        out[(size_t)orow * G4D + ocol] = c[r] + b1[ocol] + b2[ocol];
    }
}

// ---------------------------------------------------------------------------
// 2) sequential LSTM recurrence, one independent cell per block (2 blocks)
//    per step: 4 dots of length 256 against Whh rows, h kept in LDS
// ---------------------------------------------------------------------------
__global__ void __launch_bounds__(256)
ea_scan_cell(const float* __restrict__ Gxf, const float* __restrict__ Gxb,
             const float* __restrict__ Whh_f, const float* __restrict__ Whh_b,
             const float* __restrict__ h0, const float* __restrict__ c0,
             float* __restrict__ hf_all, float* __restrict__ hb_all) {
    const int cell = blockIdx.x;
    const float* Gx   = cell ? Gxb   : Gxf;
    const float* W    = cell ? Whh_b : Whh_f;
    float*       hout = cell ? hb_all : hf_all;
    __shared__ float h[DD];
    const int d = threadIdx.x;
    float c = c0[cell * DD + d];
    h[d] = h0[cell * DD + d];
    __syncthreads();
    const float* w0 = W + (size_t)d * DD;
    const float* w1 = w0 + (size_t)DD * DD;
    const float* w2 = w1 + (size_t)DD * DD;
    const float* w3 = w2 + (size_t)DD * DD;
    for (int t = 0; t < LXN; ++t) {
        const float* g = Gx + (size_t)t * G4D;
        float gi = g[d], gf = g[DD + d], gg = g[2 * DD + d], go = g[3 * DD + d];
#pragma unroll 4
        for (int k = 0; k < DD; ++k) {
            float hk = h[k];
            gi += hk * w0[k];
            gf += hk * w1[k];
            gg += hk * w2[k];
            go += hk * w3[k];
        }
        __syncthreads();           // all reads of h done
        c = sigmoidf_(gf) * c + sigmoidf_(gi) * tanhf(gg);
        float hn = sigmoidf_(go) * tanhf(c);
        h[d] = hn;
        hout[(size_t)t * DD + d] = hn;
        __syncthreads();           // new h visible
    }
}

// ---------------------------------------------------------------------------
// 3) hs = hf + hb, cast to f16 for the batched MLP GEMMs
// ---------------------------------------------------------------------------
__global__ void ea_hs(const float* __restrict__ hf_all, const float* __restrict__ hb_all,
                      _Float16* __restrict__ hs_h) {
    int i = blockIdx.x * blockDim.x + threadIdx.x;
    hs_h[i] = (_Float16)(hf_all[i] + hb_all[i]);
}

// ---------------------------------------------------------------------------
// 4) MLP layer 1 (batched over all timesteps): R1 = relu(A @ W^T + b), f16 out
// ---------------------------------------------------------------------------
__global__ void ea_mlp1_gemm(const _Float16* __restrict__ Ah, const float* __restrict__ W,
                             const float* __restrict__ bias, _Float16* __restrict__ R1) {
    const int l = threadIdx.x;
    const int m = l & 15, half = l >> 4;
    const _Float16* arow = Ah + (size_t)(blockIdx.x * 16 + m) * DD;
    const float*    bcol = W + (size_t)(blockIdx.y * 16 + m) * DD;
    v8f c = {};
#pragma unroll
    for (int ks = 0; ks < DD; ks += 32) {
        v16h a = frag_a_f16(arow, ks, half);
        v16h b = frag_b_f32(bcol, ks, half);
        c = __builtin_amdgcn_wmma_f32_16x16x32_f16(false, a, false, b, (short)0, c, false, false);
    }
#pragma unroll
    for (int r = 0; r < 8; ++r) {
        int row = blockIdx.x * 16 + r + 8 * half;
        int col = blockIdx.y * 16 + m;
        R1[(size_t)row * DD + col] = (_Float16)fmaxf(c[r] + bias[col], 0.0f);
    }
}

// ---------------------------------------------------------------------------
// 5) MLP layer 2 (batched): Out = A @ W^T + b, f32 out
// ---------------------------------------------------------------------------
__global__ void ea_mlp2_gemm(const _Float16* __restrict__ Ah, const float* __restrict__ W,
                             const float* __restrict__ bias, float* __restrict__ Out) {
    const int l = threadIdx.x;
    const int m = l & 15, half = l >> 4;
    const _Float16* arow = Ah + (size_t)(blockIdx.x * 16 + m) * DD;
    const float*    bcol = W + (size_t)(blockIdx.y * 16 + m) * DD;
    v8f c = {};
#pragma unroll
    for (int ks = 0; ks < DD; ks += 32) {
        v16h a = frag_a_f16(arow, ks, half);
        v16h b = frag_b_f32(bcol, ks, half);
        c = __builtin_amdgcn_wmma_f32_16x16x32_f16(false, a, false, b, (short)0, c, false, false);
    }
#pragma unroll
    for (int r = 0; r < 8; ++r) {
        int row = blockIdx.x * 16 + r + 8 * half;
        int col = blockIdx.y * 16 + m;
        Out[(size_t)row * DD + col] = c[r] + bias[col];
    }
}

// ---------------------------------------------------------------------------
// 6) z = mu + eps*softplus(sp); KL reduction -> acc[2]; emit z in f32 + f16
// ---------------------------------------------------------------------------
__global__ void __launch_bounds__(256)
ea_z_kl(const float* __restrict__ mu_all, const float* __restrict__ sp_all,
        const float* __restrict__ eps, float* __restrict__ zf,
        _Float16* __restrict__ zh, float* __restrict__ acc) {
    __shared__ float red[256];
    const int i = blockIdx.x * 256 + threadIdx.x;
    float mu = mu_all[i], sp = sp_all[i];
    float sg = (sp > 20.0f) ? sp : log1pf(expf(sp));   // softplus
    float z  = mu + eps[i] * sg;
    zf[i] = z;
    zh[i] = (_Float16)z;
    red[threadIdx.x] = -logf(sg) + 0.5f * (sg * sg + mu * mu) - 0.5f;
    __syncthreads();
#pragma unroll
    for (int s = 128; s > 0; s >>= 1) {
        if (threadIdx.x < s) red[threadIdx.x] += red[threadIdx.x + s];
        __syncthreads();
    }
    if (threadIdx.x == 0) atomicAdd(acc + 2, red[0]);
}

// ---------------------------------------------------------------------------
// 7) den[r] = sum_n exp( z_r . Atab[neg[n]] ), WMMA tiles + exp + row reduce
// ---------------------------------------------------------------------------
__global__ void ea_den_exp_gemm(const _Float16* __restrict__ Zh, const float* __restrict__ Atab,
                                const int* __restrict__ neg, float* __restrict__ den) {
    const int l = threadIdx.x;
    const int m = l & 15, half = l >> 4;
    const _Float16* arow = Zh + (size_t)(blockIdx.x * 16 + m) * DD;
    const float*    bcol = Atab + (size_t)neg[blockIdx.y * 16 + m] * DD;
    v8f c = {};
#pragma unroll
    for (int ks = 0; ks < DD; ks += 32) {
        v16h a = frag_a_f16(arow, ks, half);
        v16h b = frag_b_f32(bcol, ks, half);
        c = __builtin_amdgcn_wmma_f32_16x16x32_f16(false, a, false, b, (short)0, c, false, false);
    }
#pragma unroll
    for (int r = 0; r < 8; ++r) {
        float s = expf(c[r]);
        s += __shfl_xor(s, 1, 32);
        s += __shfl_xor(s, 2, 32);
        s += __shfl_xor(s, 4, 32);
        s += __shfl_xor(s, 8, 32);
        if (m == 0) atomicAdd(den + blockIdx.x * 16 + r + 8 * half, s);
    }
}

// ---------------------------------------------------------------------------
// 8) x-term: acc[0] += num_x/(num_x + den_x) per row
// ---------------------------------------------------------------------------
__global__ void ea_numx(const float* __restrict__ zf, const float* __restrict__ Ax,
                        const int* __restrict__ x, const float* __restrict__ den_x,
                        float* __restrict__ acc) {
    const int i = threadIdx.x;
    const float* zr = zf + (size_t)i * DD;
    const float* ar = Ax + (size_t)x[i] * DD;
    float s = 0.0f;
    for (int k = 0; k < DD; ++k) s += zr[k] * ar[k];
    float e = expf(s);
    atomicAdd(acc + 0, e / (e + den_x[i]));
}

// ---------------------------------------------------------------------------
// 9) y-term: acc[1] += sum_{r,n} num_y/(num_y + den_y[r])  (WMMA tiles)
// ---------------------------------------------------------------------------
__global__ void ea_yterm_gemm(const _Float16* __restrict__ Zh, const float* __restrict__ Ay,
                              const int* __restrict__ y, const float* __restrict__ den_y,
                              float* __restrict__ acc) {
    const int l = threadIdx.x;
    const int m = l & 15, half = l >> 4;
    const _Float16* arow = Zh + (size_t)(blockIdx.x * 16 + m) * DD;
    const float*    bcol = Ay + (size_t)y[blockIdx.y * 16 + m] * DD;
    v8f c = {};
#pragma unroll
    for (int ks = 0; ks < DD; ks += 32) {
        v16h a = frag_a_f16(arow, ks, half);
        v16h b = frag_b_f32(bcol, ks, half);
        c = __builtin_amdgcn_wmma_f32_16x16x32_f16(false, a, false, b, (short)0, c, false, false);
    }
    float local = 0.0f;
#pragma unroll
    for (int r = 0; r < 8; ++r) {
        int row = blockIdx.x * 16 + r + 8 * half;
        float e = expf(c[r]);
        local += e / (e + den_y[row]);
    }
    local += __shfl_xor(local, 1, 32);
    local += __shfl_xor(local, 2, 32);
    local += __shfl_xor(local, 4, 32);
    local += __shfl_xor(local, 8, 32);
    local += __shfl_xor(local, 16, 32);
    if (l == 0) atomicAdd(acc + 1, local);
}

// ---------------------------------------------------------------------------
// 10) final scalar:  out = -(log_xi + log_yi - kl)
// ---------------------------------------------------------------------------
__global__ void ea_finalize(const float* __restrict__ acc, float* __restrict__ out) {
    out[0] = -(acc[0] + acc[1] * (1.0f / (float)LXN) - acc[2]);
}

// ---------------------------------------------------------------------------
extern "C" void kernel_launch(void* const* d_in, const int* in_sizes, int n_in,
                              void* d_out, int out_size, void* d_ws, size_t ws_size,
                              hipStream_t stream) {
    (void)in_sizes; (void)n_in; (void)out_size; (void)ws_size;
    const int*   x     = (const int*)d_in[0];
    const int*   y     = (const int*)d_in[1];
    const int*   neg_x = (const int*)d_in[2];
    const int*   neg_y = (const int*)d_in[3];
    const float* eps   = (const float*)d_in[4];
    const float* h0    = (const float*)d_in[5];
    const float* c0    = (const float*)d_in[6];
    const float* E_x   = (const float*)d_in[7];
    const float* Wih_f = (const float*)d_in[8];
    const float* Whh_f = (const float*)d_in[9];
    const float* bih_f = (const float*)d_in[10];
    const float* bhh_f = (const float*)d_in[11];
    const float* Wih_b = (const float*)d_in[12];
    const float* Whh_b = (const float*)d_in[13];
    const float* bih_b = (const float*)d_in[14];
    const float* bhh_b = (const float*)d_in[15];
    const float* Wmu1  = (const float*)d_in[16];
    const float* bmu1  = (const float*)d_in[17];
    const float* Wmu2  = (const float*)d_in[18];
    const float* bmu2  = (const float*)d_in[19];
    const float* Wsg1  = (const float*)d_in[20];
    const float* bsg1  = (const float*)d_in[21];
    const float* Wsg2  = (const float*)d_in[22];
    const float* bsg2  = (const float*)d_in[23];
    const float* Ax    = (const float*)d_in[24];
    const float* Ay    = (const float*)d_in[25];
    float* out = (float*)d_out;

    // workspace layout (float units)
    float* ws     = (float*)d_ws;
    float* zf     = ws;                                  // 65536
    float* Gxf    = ws + 65536;                          // 262144
    float* Gxb    = ws + 65536 + 262144;                 // 262144
    float* hf_all = ws + 589824;                         // 65536
    float* hb_all = ws + 655360;                         // 65536
    float* mu_all = ws + 720896;                         // 65536
    float* sp_all = ws + 786432;                         // 65536
    float* den_x  = ws + 851968;                         // 256
    float* den_y  = ws + 852224;                         // 256
    float* acc    = ws + 852480;                         // 16
    _Float16* zh   = (_Float16*)(ws + 852496);           // 65536 halves
    _Float16* hs_h = (_Float16*)(ws + 885264);           // 65536 halves
    _Float16* r1m  = (_Float16*)(ws + 918032);           // 65536 halves
    _Float16* r1s  = (_Float16*)(ws + 950800);           // 65536 halves

    ea_init<<<1, 256, 0, stream>>>(den_x, den_y, acc);

    // Gate-input precompute GEMMs: [256 x 1024] = E_x[x] @ Wih^T + bih + bhh
    ea_gx_gemm<<<dim3(LXN / 16, G4D / 16), 32, 0, stream>>>(E_x, x, Wih_f, bih_f, bhh_f, Gxf);
    ea_gx_gemm<<<dim3(LXN / 16, G4D / 16), 32, 0, stream>>>(E_x, x, Wih_b, bih_b, bhh_b, Gxb);

    // Two independent recurrences, concurrently on two WGPs
    ea_scan_cell<<<2, 256, 0, stream>>>(Gxf, Gxb, Whh_f, Whh_b, h0, c0, hf_all, hb_all);

    // hs = hf + hb (f16), then batched MLP heads as WMMA GEMMs
    ea_hs<<<LXN, 256, 0, stream>>>(hf_all, hb_all, hs_h);
    ea_mlp1_gemm<<<dim3(LXN / 16, DD / 16), 32, 0, stream>>>(hs_h, Wmu1, bmu1, r1m);
    ea_mlp1_gemm<<<dim3(LXN / 16, DD / 16), 32, 0, stream>>>(hs_h, Wsg1, bsg1, r1s);
    ea_mlp2_gemm<<<dim3(LXN / 16, DD / 16), 32, 0, stream>>>(r1m, Wmu2, bmu2, mu_all);
    ea_mlp2_gemm<<<dim3(LXN / 16, DD / 16), 32, 0, stream>>>(r1s, Wsg2, bsg2, sp_all);

    // z, KL
    ea_z_kl<<<LXN, 256, 0, stream>>>(mu_all, sp_all, eps, zf, zh, acc);

    // den_x / den_y : sum over 8192 negatives of exp(z . A[neg])
    ea_den_exp_gemm<<<dim3(LXN / 16, NNEG / 16), 32, 0, stream>>>(zh, Ax, neg_x, den_x);
    ea_den_exp_gemm<<<dim3(LXN / 16, NNEG / 16), 32, 0, stream>>>(zh, Ay, neg_y, den_y);

    // x-term (needs den_x), y-term (needs den_y)
    ea_numx<<<1, 256, 0, stream>>>(zf, Ax, x, den_x, acc);
    ea_yterm_gemm<<<dim3(LXN / 16, LXN / 16), 32, 0, stream>>>(zh, Ay, y, den_y, acc);

    // combine
    ea_finalize<<<1, 1, 0, stream>>>(acc, out);
}
